// LlamaLolcatsAttentionActual_61065845014941
// MI455X (gfx1250) — matrix operand order repaired and checked by reading, not compiled, not run on hardware
//
#include <hip/hip_runtime.h>
#include <hip/hip_bf16.h>
#include <cstdint>

// ---------------------------------------------------------------------------
// LoLCATs linearized attention for MI455X (gfx1250, wave32, WMMA bf16).
// T=2048, H=32, KVH=8, D=128, F=64 (->128 after two-sided softmax), W=64.
// All LDS operands laid out so fragment loads are ds_load_b128 pairs.
// ---------------------------------------------------------------------------

typedef __bf16 bf16_t;
typedef __attribute__((ext_vector_type(16))) __bf16 v16bf;
typedef __attribute__((ext_vector_type(8)))  __bf16 v8bf;
typedef __attribute__((ext_vector_type(8)))  float  v8f;

#define T_LEN 2048
#define NH    32
#define NKV   8
#define HD    128
#define FD    64
#define CHUNK 64
#define NCH   32    // T / CHUNK

// padded LDS row strides (elements); all multiples of 16 bytes (16B alignment for
// b128 loads), and odd multiples of 4 DWORDs to spread LDS banks.
#define SW    136   // rows of 128 bf16
#define SC72  72    // rows of 64 bf16
#define SZ    68    // rows of 64 f32
#define SSC   132   // rows of 128 f32

__device__ __forceinline__ v8f wmma_bf16(v16bf a, v16bf b, v8f c) {
  return __builtin_amdgcn_wmma_f32_16x16x32_bf16(
      false, a, false, b, (short)0, c, false, false);
}

// A fragment (16x32), row-major source, row stride rs (mult of 8).
// Lane l, group hi: frag[0..7] = A(row0+l, k0+hi*8 .. +7),
//                   frag[8..15] = A(row0+l, k0+16+hi*8 .. +7).
__device__ __forceinline__ v16bf load_a_rm(const bf16_t* p, int rs,
                                           int row0, int k0) {
  int lane = threadIdx.x & 31;
  int l = lane & 15, hi = lane >> 4;
  const bf16_t* base = p + (row0 + l) * rs + k0 + hi * 8;
  v8bf lo = *(const v8bf*)(base);
  v8bf hh = *(const v8bf*)(base + 16);
  v16bf f;
#pragma unroll
  for (int i = 0; i < 8; ++i) { f[i] = lo[i]; f[8 + i] = hh[i]; }
  return f;
}

// B fragment (32x16), K-contiguous source: element (k,n) = p[(n)*rs + k].
// Lane l, group hi: frag[0..7] = B(k0+hi*8 .. +7, n0+l),
//                   frag[8..15] = B(k0+16+hi*8 .. +7, n0+l).
__device__ __forceinline__ v16bf load_b_kc(const bf16_t* p, int rs,
                                           int k0, int n0) {
  int lane = threadIdx.x & 31;
  int l = lane & 15, hi = lane >> 4;
  const bf16_t* base = p + (n0 + l) * rs + k0 + hi * 8;
  v8bf lo = *(const v8bf*)(base);
  v8bf hh = *(const v8bf*)(base + 16);
  v16bf f;
#pragma unroll
  for (int i = 0; i < 8; ++i) { f[i] = lo[i]; f[8 + i] = hh[i]; }
  return f;
}

// ---------------------------------------------------------------------------
// Kernel 1: feature maps.  z = q(64x128)*W(128x64) via WMMA; two-sided
// softmax over 64 features; clipped concat -> bf16 f_q/f_k.
// grid (NH, NCH), 256 threads, ~100KB dynamic LDS.
// ---------------------------------------------------------------------------
__global__ __launch_bounds__(256) void featmap_kernel(
    const float* __restrict__ q_g, const float* __restrict__ k_g,
    const float* __restrict__ wq_g, const float* __restrict__ wk_g,
    bf16_t* __restrict__ fq_g, bf16_t* __restrict__ fk_g) {
  int h = blockIdx.x, c = blockIdx.y;
  extern __shared__ char smem[];
  bf16_t* qs  = (bf16_t*)smem;          // [64][128] row-major (A)
  bf16_t* ks  = qs + CHUNK * HD;        // [64][128]
  bf16_t* wqt = ks + CHUNK * HD;        // [64 f][SW] K(d)-contiguous (B)
  bf16_t* wkt = wqt + FD * SW;          // [64 f][SW]
  float*  zq  = (float*)(wkt + FD * SW);  // [64][SZ]
  float*  zk  = zq + CHUNK * SZ;          // [64][SZ]

  int tid = threadIdx.x;
  int kvh = h >> 2;
  for (int idx = tid; idx < CHUNK * HD; idx += 256) {
    int t = c * CHUNK + (idx >> 7);
    int d = idx & 127;
    qs[idx] = (bf16_t)q_g[t * (NH * HD) + h * HD + d];
    ks[idx] = (bf16_t)k_g[t * (NKV * HD) + kvh * HD + d];
  }
  for (int idx = tid; idx < HD * FD; idx += 256) {
    int d = idx >> 6, f = idx & 63;                  // global (d,f) row-major
    wqt[f * SW + d] = (bf16_t)wq_g[h * HD * FD + idx];
    wkt[f * SW + d] = (bf16_t)wk_g[h * HD * FD + idx];
  }
  __syncthreads();

  int wv = tid >> 5;
  int lane = tid & 31, l = lane & 15, hi = lane >> 4;
  const bf16_t* src = (wv < 4) ? qs : ks;
  const bf16_t* wsl = (wv < 4) ? wqt : wkt;
  float* z = (wv < 4) ? zq : zk;
  int rt = wv & 3;  // 16-token row tile
#pragma unroll
  for (int ct = 0; ct < 4; ++ct) {
    v8f acc = {};
#pragma unroll
    for (int kk = 0; kk < 4; ++kk) {
      v16bf a = load_a_rm(src, HD, rt * 16, kk * 32);
      v16bf b = load_b_kc(wsl, SW, kk * 32, ct * 16);
      acc = wmma_bf16(a, b, acc);
    }
#pragma unroll
    for (int r = 0; r < 8; ++r)
      z[(rt * 16 + r + 8 * hi) * SZ + ct * 16 + l] = acc[r];
  }
  __syncthreads();

  if (tid < 128) {
    float* z2 = (tid < 64) ? zq : zk;
    bf16_t* out = (tid < 64) ? fq_g : fk_g;
    int m = tid & 63;
    float mx = -3.0e38f, mn = 3.0e38f;
    for (int j = 0; j < FD; ++j) {
      float v = z2[m * SZ + j];
      mx = fmaxf(mx, v); mn = fminf(mn, v);
    }
    float sp = 0.f, sn = 0.f;
    for (int j = 0; j < FD; ++j) {
      float v = z2[m * SZ + j];
      sp += __expf(v - mx);        // softmax(z):  max = mx
      sn += __expf(mn - v);        // softmax(-z): max(-z) = -mn
    }
    int t = c * CHUNK + m;
    bf16_t* o = out + ((size_t)h * T_LEN + t) * HD;
    for (int j = 0; j < FD; ++j) {
      float v = z2[m * SZ + j];
      o[j]      = (bf16_t)fmaxf(__expf(v - mx) / sp, 1e-12f);
      o[FD + j] = (bf16_t)fmaxf(__expf(mn - v) / sn, 1e-12f);
    }
  }
}

// ---------------------------------------------------------------------------
// Kernel 2: per-chunk KV state O_c = f_k^T(128x64)*v(64x128) + key sums.
// grid (NH, NCH), 256 threads.  Both operands staged token(K)-contiguous.
// ---------------------------------------------------------------------------
__global__ __launch_bounds__(256) void chunkstate_kernel(
    const bf16_t* __restrict__ fk_g, const float* __restrict__ v_g,
    float* __restrict__ O_g, float* __restrict__ ks_g) {
  int h = blockIdx.x, c = blockIdx.y;
  __shared__ bf16_t fkt[HD * SC72];   // [128 f][72] : A row f, K=t contiguous
  __shared__ bf16_t vt[HD * SC72];    // [128 d][72] : B col d, K=t contiguous
  int tid = threadIdx.x;
  int kvh = h >> 2;
  for (int idx = tid; idx < CHUNK * HD; idx += 256) {
    int t = c * CHUNK + (idx >> 7);
    int tl = idx >> 7, d = idx & 127;
    fkt[d * SC72 + tl] = fk_g[((size_t)h * T_LEN + t) * HD + d];
    vt[d * SC72 + tl]  = (bf16_t)v_g[t * (NKV * HD) + kvh * HD + d];
  }
  __syncthreads();

  int wv = tid >> 5, lane = tid & 31, l = lane & 15, hi = lane >> 4;
  float* Ob = O_g + ((size_t)h * NCH + c) * HD * HD;
#pragma unroll
  for (int nt = 0; nt < 8; ++nt) {
    v8f acc = {};
#pragma unroll
    for (int kk = 0; kk < 2; ++kk) {
      v16bf a = load_a_rm(fkt, SC72, wv * 16, kk * 32);
      v16bf b = load_b_kc(vt, SC72, kk * 32, nt * 16);
      acc = wmma_bf16(a, b, acc);
    }
#pragma unroll
    for (int r = 0; r < 8; ++r)
      Ob[(wv * 16 + r + 8 * hi) * HD + nt * 16 + l] = acc[r];
  }
  if (tid < HD) {
    float s = 0.f;
    for (int t = 0; t < CHUNK; ++t) s += (float)fkt[tid * SC72 + t];
    ks_g[((size_t)h * NCH + c) * HD + tid] = s;
  }
}

// ---------------------------------------------------------------------------
// Kernel 3/4: exclusive-by-2 prefix over chunk states (f32 accumulate,
// bf16 out for the WMMA consumer) and over key-feature sums.
// ---------------------------------------------------------------------------
__global__ void prefix_kernel(const float* __restrict__ O_g,
                              bf16_t* __restrict__ Sp_g) {
  int h = blockIdx.x;
  int e = blockIdx.y * 256 + threadIdx.x;          // 0..16383
  const float* Ob = O_g + (size_t)h * NCH * HD * HD;
  bf16_t* Sb = Sp_g + (size_t)h * NCH * HD * HD;
  float run = 0.f;
  for (int c = 0; c < NCH; ++c) {
    Sb[(size_t)c * HD * HD + e] = (bf16_t)run;     // Sum_{j<=c-2} O_j
    if (c >= 1) run += Ob[(size_t)(c - 1) * HD * HD + e];
  }
}

__global__ void ksprefix_kernel(const float* __restrict__ ks_g,
                                float* __restrict__ ksp_g) {
  int h = blockIdx.x;
  int f = threadIdx.x;                              // 128 threads
  const float* a = ks_g + (size_t)h * NCH * HD;
  float* b = ksp_g + (size_t)h * NCH * HD;
  float run = 0.f;
  for (int c = 0; c < NCH; ++c) {
    b[c * HD + f] = run;
    if (c >= 1) run += a[(c - 1) * HD + f];
  }
}

// ---------------------------------------------------------------------------
// Kernel 5: output.  Window scores q*k_win^T (WMMA), masked softmax rows,
// fused y = probs*v_win + f_q*S_pref (WMMA), divide by sum_sm + sum_ln.
// grid (NH, NCH), 256 threads, ~170KB dynamic LDS.
// ---------------------------------------------------------------------------
__global__ __launch_bounds__(256) void output_kernel(
    const float* __restrict__ q_g, const float* __restrict__ k_g,
    const float* __restrict__ v_g, const bf16_t* __restrict__ fq_g,
    const bf16_t* __restrict__ Sp_g, const float* __restrict__ ksp_g,
    const float* __restrict__ wf_g, float* __restrict__ out) {
  int h = blockIdx.x, c = blockIdx.y;
  extern __shared__ char smem[];
  bf16_t* qs  = (bf16_t*)smem;            // [64][SW] A; reused as probs later
  bf16_t* fqs = qs + CHUNK * SW;          // [64][SW] A + scalar row reads
  bf16_t* kw  = fqs + CHUNK * SW;         // [128 j][SW] B for QK^T (K=d contig)
  bf16_t* vwt = kw + 2 * CHUNK * SW;      // [128 d][SW] B for PV  (K=j contig)
  bf16_t* spt = vwt + 2 * CHUNK * SW;     // [128 d][SW] B for f_q*S (K=f contig)
  float*  sc  = (float*)(spt + 2 * CHUNK * SW);  // [64][SSC] scores f32
  float*  kspl = sc + CHUNK * SSC;        // 128
  float*  den  = kspl + HD;               // 64
  bf16_t* pb   = qs;                      // probs alias (q dead after scores)

  int tid = threadIdx.x;
  int kvh = h >> 2;
  for (int idx = tid; idx < CHUNK * HD; idx += 256) {
    int t = c * CHUNK + (idx >> 7);
    int tl = idx >> 7, d = idx & 127;
    qs[tl * SW + d]  = (bf16_t)q_g[t * (NH * HD) + h * HD + d];
    fqs[tl * SW + d] = fq_g[((size_t)h * T_LEN + t) * HD + d];
  }
  for (int idx = tid; idx < 2 * CHUNK * HD; idx += 256) {
    int j = idx >> 7, d = idx & 127;
    int gk = (c - 1) * CHUNK + j;                  // global key index
    if (gk >= 0) {
      kw[j * SW + d]  = (bf16_t)k_g[gk * (NKV * HD) + kvh * HD + d];
      vwt[d * SW + j] = (bf16_t)v_g[gk * (NKV * HD) + kvh * HD + d];
    } else {
      kw[j * SW + d]  = (bf16_t)0.f;
      vwt[d * SW + j] = (bf16_t)0.f;
    }
    // Sp_g is [f][d] row-major; store K(f)-contiguous per output column d.
    spt[d * SW + j] = Sp_g[((size_t)h * NCH + c) * HD * HD + j * HD + d];
  }
  if (tid < HD) kspl[tid] = ksp_g[((size_t)h * NCH + c) * HD + tid];
  __syncthreads();

  int wv = tid >> 5, lane = tid & 31, l = lane & 15, hi = lane >> 4;
  const float scale = 0.08838834764831845f;        // 1/sqrt(128)

  // --- window scores: (64x128) = q(64x128d) * kw^T ---
#pragma unroll
  for (int t4 = 0; t4 < 4; ++t4) {
    int tile = wv * 4 + t4;                        // 32 tiles (4 rows x 8 cols)
    int row = tile >> 3, col = tile & 7;
    v8f acc = {};
#pragma unroll
    for (int kk = 0; kk < 4; ++kk) {
      v16bf a = load_a_rm(qs, SW, row * 16, kk * 32);
      v16bf b = load_b_kc(kw, SW, kk * 32, col * 16);
      acc = wmma_bf16(a, b, acc);
    }
#pragma unroll
    for (int r = 0; r < 8; ++r)
      sc[(row * 16 + r + 8 * hi) * SSC + col * 16 + l] = acc[r] * scale;
  }
  __syncthreads();

  // --- masked softmax rows + denominators (64 rows, one thread each) ---
  if (tid < CHUNK) {
    int m = tid;
    float wf = 1.f / (1.f + __expf(-wf_g[h]));     // sigmoid(window_factor)
    int jstart = (c == 0) ? 64 : 0;                // no previous chunk at c=0
    float mx = -3.0e38f;
    for (int j = jstart; j < 128; ++j) {
      bool ok = (j < 64) || ((j - 64) <= m);       // prev chunk full, self tril
      if (ok) mx = fmaxf(mx, sc[m * SSC + j]);
    }
    float ssum = 0.f;
    for (int j = 0; j < 128; ++j) {
      bool ok = (j >= jstart) && ((j < 64) || ((j - 64) <= m));
      float p = ok ? (wf * __expf(sc[m * SSC + j] - mx)) : 0.f;
      pb[m * SW + j] = (bf16_t)p;
      ssum += p;
    }
    float sl = 0.f;
    for (int f = 0; f < HD; ++f) sl += (float)fqs[m * SW + f] * kspl[f];
    den[m] = ssum + sl;
  }
  __syncthreads();

  // --- y = probs * v_win + f_q * S_pref, then divide by denominator ---
#pragma unroll
  for (int t4 = 0; t4 < 4; ++t4) {
    int tile = wv * 4 + t4;
    int row = tile >> 3, col = tile & 7;
    v8f acc = {};
#pragma unroll
    for (int kk = 0; kk < 4; ++kk) {
      v16bf a = load_a_rm(pb, SW, row * 16, kk * 32);
      v16bf b = load_b_kc(vwt, SW, kk * 32, col * 16);
      acc = wmma_bf16(a, b, acc);
    }
#pragma unroll
    for (int kk = 0; kk < 4; ++kk) {
      v16bf a = load_a_rm(fqs, SW, row * 16, kk * 32);
      v16bf b = load_b_kc(spt, SW, kk * 32, col * 16);
      acc = wmma_bf16(a, b, acc);
    }
#pragma unroll
    for (int r = 0; r < 8; ++r) {
      int m = row * 16 + r + 8 * hi;
      int d = col * 16 + l;
      out[((size_t)h * T_LEN + c * CHUNK + m) * HD + d] = acc[r] / den[m];
    }
  }
}

// ---------------------------------------------------------------------------
// Launch: 5 kernels on `stream`.  Workspace (~130 MB from d_ws):
//   fq, fk (bf16, 16MB ea), O (f32, 64MB), Sp (bf16, 32MB), ks/ksp (f32).
// ---------------------------------------------------------------------------
extern "C" void kernel_launch(void* const* d_in, const int* in_sizes, int n_in,
                              void* d_out, int out_size, void* d_ws, size_t ws_size,
                              hipStream_t stream) {
  const float* q    = (const float*)d_in[0];
  const float* k    = (const float*)d_in[1];
  const float* v    = (const float*)d_in[2];
  const float* wq   = (const float*)d_in[3];
  const float* wk   = (const float*)d_in[4];
  const float* wfac = (const float*)d_in[5];
  float* out = (float*)d_out;

  char* ws = (char*)d_ws;
  auto carve = [&](size_t bytes) {
    char* p = ws;
    ws += (bytes + 255) & ~(size_t)255;
    return p;
  };
  bf16_t* fq  = (bf16_t*)carve(sizeof(bf16_t) * (size_t)NH * T_LEN * HD);
  bf16_t* fk  = (bf16_t*)carve(sizeof(bf16_t) * (size_t)NH * T_LEN * HD);
  float*  O   = (float*) carve(sizeof(float)  * (size_t)NH * NCH * HD * HD);
  bf16_t* Sp  = (bf16_t*)carve(sizeof(bf16_t) * (size_t)NH * NCH * HD * HD);
  float*  ks  = (float*) carve(sizeof(float)  * (size_t)NH * NCH * HD);
  float*  ksp = (float*) carve(sizeof(float)  * (size_t)NH * NCH * HD);

  dim3 gridHC(NH, NCH);

  size_t sh1 = (size_t)(2 * CHUNK * HD + 2 * FD * SW) * sizeof(bf16_t)
             + (size_t)(2 * CHUNK * SZ) * sizeof(float);
  featmap_kernel<<<gridHC, 256, sh1, stream>>>(q, k, wq, wk, fq, fk);

  chunkstate_kernel<<<gridHC, 256, 0, stream>>>(fk, v, O, ks);

  prefix_kernel<<<dim3(NH, (HD * HD) / 256), 256, 0, stream>>>(O, Sp);
  ksprefix_kernel<<<NH, HD, 0, stream>>>(ks, ksp);

  size_t sh5 = (size_t)(2 * CHUNK * SW + 3 * 2 * CHUNK * SW) * sizeof(bf16_t)
             + (size_t)(CHUNK * SSC + HD + CHUNK) * sizeof(float);
  output_kernel<<<gridHC, 256, sh5, stream>>>(q, k, v, fq, Sp, ksp, wfac, out);
}
